// DisentangledRelationalCrossAttentionV2_51230369907057
// MI455X (gfx1250) — compile-verified, matrix operand order checked by reading, not run on hardware
//
#include <hip/hip_runtime.h>
#include <hip/hip_bf16.h>

typedef __bf16 bf16_t;
typedef __attribute__((ext_vector_type(16))) __bf16 v16bf;
typedef __attribute__((ext_vector_type(8)))  __bf16 v8bf;
typedef __attribute__((ext_vector_type(8)))  float  v8f;

#define L_SEQ 512
#define DMODEL 1024
#define NHEAD 16
#define NKVH 4
#define NRH 16
#define HDIM 64
#define ATTN_SCALE 0.125f
#define REL_SCALE 0.125f

static __device__ __forceinline__ bf16_t f2bf(float f) {
  unsigned int u = __builtin_bit_cast(unsigned int, f);
  unsigned int r = u + 0x7FFFu + ((u >> 16) & 1u);
  unsigned short hs = (unsigned short)(r >> 16);
  return __builtin_bit_cast(bf16_t, hs);
}

// ---------------- elementwise converts ----------------
__global__ void k_f32_to_bf16(const float* __restrict__ in, bf16_t* __restrict__ out, int n) {
  int i = blockIdx.x * 256 + threadIdx.x;
  if (i < n) out[i] = f2bf(in[i]);
}

// in: [K][N] f32 row-major  ->  out: [N][K] bf16 row-major
__global__ void k_transpose_to_bf16(const float* __restrict__ in, bf16_t* __restrict__ out,
                                    int K, int N) {
  int i = blockIdx.x * 256 + threadIdx.x;
  if (i < N * K) {
    int n = i / K, k = i % K;
    out[i] = f2bf(in[(long)k * N + n]);
  }
}

// ---------------- generic batched WMMA GEMM ----------------
// C[z][m][n] = scale * sum_k A[z][m][k] * BT[z][n][k]   (bf16 in, f32 accum)
// One wave (32 threads) computes a 32x32 tile via 2x2 v_wmma_f32_16x16x32_bf16.
static __device__ __forceinline__ v16bf load_frag(const bf16_t* p) {
  // A-fragment layout (16-bit, 16x32): lanes 0-15 hold K 0..7 & 16..23,
  // lanes 16-31 hold K 8..15 & 24..31 (kb offset applied by caller).
  v8bf lo = *reinterpret_cast<const v8bf*>(p);
  v8bf hi = *reinterpret_cast<const v8bf*>(p + 16);
  union { v16bf v; v8bf h[2]; } u;
  u.h[0] = lo; u.h[1] = hi;
  return u.v;
}

__global__ void k_gemm_bf16(const bf16_t* __restrict__ A, int ldA, long sAz,
                            const bf16_t* __restrict__ BT, int ldBT, long sBTz,
                            float* __restrict__ Cf, bf16_t* __restrict__ Cb,
                            long cm, long cn, long cz,
                            int K, float scale) {
  const int lane = threadIdx.x;
  const int lrow = lane & 15;
  const int lhi  = lane >> 4;
  const int z = blockIdx.z;
  const int rowBase = blockIdx.y * 32;
  const int colBase = blockIdx.x * 32;
  A  += (long)z * sAz;
  BT += (long)z * sBTz;

  v8f acc00 = {}, acc01 = {}, acc10 = {}, acc11 = {};
  for (int ko = 0; ko < K; ko += 32) {
    const int kb = ko + lhi * 8;
    v16bf a0 = load_frag(A + (long)(rowBase + lrow)      * ldA + kb);
    v16bf a1 = load_frag(A + (long)(rowBase + 16 + lrow) * ldA + kb);
    v16bf b0 = load_frag(BT + (long)(colBase + lrow)      * ldBT + kb);
    v16bf b1 = load_frag(BT + (long)(colBase + 16 + lrow) * ldBT + kb);
    acc00 = __builtin_amdgcn_wmma_f32_16x16x32_bf16(false, a0, false, b0, (short)0, acc00, false, false);
    acc01 = __builtin_amdgcn_wmma_f32_16x16x32_bf16(false, a0, false, b1, (short)0, acc01, false, false);
    acc10 = __builtin_amdgcn_wmma_f32_16x16x32_bf16(false, a1, false, b0, (short)0, acc10, false, false);
    acc11 = __builtin_amdgcn_wmma_f32_16x16x32_bf16(false, a1, false, b1, (short)0, acc11, false, false);
  }

  // C 16x16 f32 layout: VGPR v holds M=v (lanes 0-15) / M=v+8 (lanes 16-31), N=lane%16
  const long zoff = (long)z * cz;
#pragma unroll
  for (int v = 0; v < 8; ++v) {
    long m0 = rowBase + v + 8 * lhi;
    long m1 = m0 + 16;
    long n0 = colBase + lrow;
    long n1 = n0 + 16;
    float v00 = acc00[v] * scale, v01 = acc01[v] * scale;
    float v10 = acc10[v] * scale, v11 = acc11[v] * scale;
    long i00 = zoff + m0 * cm + n0 * cn;
    long i01 = zoff + m0 * cm + n1 * cn;
    long i10 = zoff + m1 * cm + n0 * cn;
    long i11 = zoff + m1 * cm + n1 * cn;
    if (Cf) { Cf[i00] = v00; Cf[i01] = v01; Cf[i10] = v10; Cf[i11] = v11; }
    if (Cb) { Cb[i00] = f2bf(v00); Cb[i01] = f2bf(v01); Cb[i10] = f2bf(v10); Cb[i11] = f2bf(v11); }
  }
}

// ---------------- RoPE ----------------
__global__ void k_rope_q(const float* __restrict__ qf, const float* __restrict__ fc,
                         const float* __restrict__ fs, bf16_t* __restrict__ qb) {
  int p = blockIdx.x * 256 + threadIdx.x;          // L*NH*32 pairs
  if (p >= L_SEQ * NHEAD * 32) return;
  int i = p >> 9; int rem = p & 511;
  int hh = rem >> 5; int t = rem & 31;
  float c = fc[i * 32 + t], s = fs[i * 32 + t];
  long base = (long)i * DMODEL + hh * HDIM + 2 * t;
  float re = qf[base], im = qf[base + 1];
  qb[base]     = f2bf(re * c - im * s);
  qb[base + 1] = f2bf(re * s + im * c);
}

// RoPE on k (4 kv heads) + GQA replicate to 16 heads, bf16 out [L][1024]
__global__ void k_rope_k_rep(const float* __restrict__ kf, const float* __restrict__ fc,
                             const float* __restrict__ fs, bf16_t* __restrict__ kb) {
  int p = blockIdx.x * 256 + threadIdx.x;          // L*NKV*32 pairs
  if (p >= L_SEQ * NKVH * 32) return;
  int i = p >> 7; int rem = p & 127;
  int kv = rem >> 5; int t = rem & 31;
  float c = fc[i * 32 + t], s = fs[i * 32 + t];
  long sb = (long)i * (NKVH * HDIM) + kv * HDIM + 2 * t;
  float re = kf[sb], im = kf[sb + 1];
  bf16_t b0 = f2bf(re * c - im * s);
  bf16_t b1 = f2bf(re * s + im * c);
#pragma unroll
  for (int rr = 0; rr < 4; ++rr) {
    long db = (long)i * DMODEL + (kv * 4 + rr) * HDIM + 2 * t;
    kb[db] = b0; kb[db + 1] = b1;
  }
}

// sv f32 [L][NKV*64] -> svT bf16 [NH][64][L]  (transposed + replicated)
__global__ void k_svt_rep(const float* __restrict__ svf, bf16_t* __restrict__ svT) {
  int idx = blockIdx.x * 256 + threadIdx.x;        // NH*64*L
  if (idx >= NHEAD * HDIM * L_SEQ) return;
  int h = idx >> 15; int rem = idx & 32767;
  int d = rem >> 9; int j = rem & 511;
  svT[idx] = f2bf(svf[(long)j * (NKVH * HDIM) + (h >> 2) * HDIM + d]);
}

// ---------------- causal softmax (in-place on d_out attn region) ----------------
__global__ void k_softmax(float* __restrict__ attn, bf16_t* __restrict__ attnb) {
  __shared__ float red[256];
  int row = blockIdx.x;                            // h*512+i
  int i = row & 511;
  float* sp = attn + (long)row * L_SEQ;
  int tid = threadIdx.x;
  float s0 = sp[tid], s1 = sp[tid + 256];
  bool ok0 = tid <= i, ok1 = (tid + 256) <= i;
  float m = -3.4e38f;
  if (ok0) m = s0;
  if (ok1 && s1 > m) m = s1;
  red[tid] = m; __syncthreads();
  for (int st = 128; st > 0; st >>= 1) {
    if (tid < st) { float o = red[tid + st]; if (o > red[tid]) red[tid] = o; }
    __syncthreads();
  }
  float mx = red[0]; __syncthreads();
  float e0 = ok0 ? __expf(s0 - mx) : 0.f;
  float e1 = ok1 ? __expf(s1 - mx) : 0.f;
  red[tid] = e0 + e1; __syncthreads();
  for (int st = 128; st > 0; st >>= 1) {
    if (tid < st) red[tid] += red[tid + st];
    __syncthreads();
  }
  float inv = 1.f / red[0];
  float a0 = e0 * inv, a1 = e1 * inv;
  sp[tid] = a0; sp[tid + 256] = a1;
  bf16_t* bp = attnb + (long)row * L_SEQ;
  bp[tid] = f2bf(a0); bp[tid + 256] = f2bf(a1);
}

// AR[i][h][r] = sum_j attn[h][i][j] * relations[i][j][r]
__global__ void k_ar(const float* __restrict__ attn, const float* __restrict__ rel,
                     float* __restrict__ AR) {
  __shared__ float red[256];
  int b = blockIdx.x;                              // NH*L blocks
  int h = b >> 9, i = b & 511;
  int tid = threadIdx.x;
  int r = tid & 15, g = tid >> 4;
  const float* ap = attn + ((long)h * L_SEQ + i) * L_SEQ;
  const float* rp = rel + (long)i * L_SEQ * NRH;
  float acc = 0.f;
  for (int j = g; j < L_SEQ; j += 16)
    acc += ap[j] * rp[(long)j * NRH + r];
  red[tid] = acc; __syncthreads();
  for (int st = 8; st > 0; st >>= 1) {
    if (g < st) red[tid] += red[(g + st) * 16 + r];
    __syncthreads();
  }
  if (g == 0) AR[((long)i * NHEAD + h) * NRH + r] = red[r];
}

// Y[i][c] = AS[i][c] + sum_r AR[i][c/64][r] * wr[r][c]   -> bf16
__global__ void k_epilogue(const float* __restrict__ AS, const float* __restrict__ AR,
                           const float* __restrict__ wr, bf16_t* __restrict__ Yb) {
  int idx = blockIdx.x * 256 + threadIdx.x;        // L*DMODEL
  if (idx >= L_SEQ * DMODEL) return;
  int i = idx >> 10, c = idx & 1023, h = c >> 6;
  float acc = AS[idx];
  const float* arp = AR + (long)i * (NHEAD * NRH) + h * NRH;
#pragma unroll
  for (int r = 0; r < NRH; ++r) acc += arp[r] * wr[r * DMODEL + c];
  Yb[idx] = f2bf(acc);
}

// ---------------- host launch ----------------
extern "C" void kernel_launch(void* const* d_in, const int* in_sizes, int n_in,
                              void* d_out, int out_size, void* d_ws, size_t ws_size,
                              hipStream_t stream) {
  (void)in_sizes; (void)n_in; (void)out_size; (void)ws_size;
  const float* x   = (const float*)d_in[0];
  const float* sym = (const float*)d_in[1];
  const float* fc  = (const float*)d_in[2];
  const float* fs  = (const float*)d_in[3];
  const float* wq  = (const float*)d_in[4];
  const float* wk  = (const float*)d_in[5];
  const float* wqr = (const float*)d_in[6];
  const float* wkr = (const float*)d_in[7];
  const float* wr  = (const float*)d_in[8];
  const float* wv  = (const float*)d_in[9];
  const float* wo  = (const float*)d_in[10];

  float* out  = (float*)d_out;                       // [512][1024]
  float* attn = out + (long)L_SEQ * DMODEL;          // [16][512][512]
  float* rel  = attn + (long)NHEAD * L_SEQ * L_SEQ;  // [512][512][16]

  char* w = (char*)d_ws;
  auto alloc = [&](size_t bytes) { char* p = w; w += (bytes + 255) & ~(size_t)255; return p; };
  bf16_t* xb   = (bf16_t*)alloc((size_t)L_SEQ * DMODEL * 2);
  bf16_t* symb = (bf16_t*)alloc((size_t)L_SEQ * DMODEL * 2);
  bf16_t* wqT  = (bf16_t*)alloc((size_t)DMODEL * DMODEL * 2);
  bf16_t* wkT  = (bf16_t*)alloc((size_t)DMODEL * 256 * 2);
  bf16_t* wqrT = (bf16_t*)alloc((size_t)DMODEL * DMODEL * 2);
  bf16_t* wkrT = (bf16_t*)alloc((size_t)DMODEL * DMODEL * 2);
  bf16_t* wvT  = (bf16_t*)alloc((size_t)DMODEL * 256 * 2);
  bf16_t* woT  = (bf16_t*)alloc((size_t)DMODEL * DMODEL * 2);
  float*  qf   = (float*)alloc((size_t)L_SEQ * DMODEL * 4);
  float*  kf   = (float*)alloc((size_t)L_SEQ * 256 * 4);
  float*  svf  = (float*)alloc((size_t)L_SEQ * 256 * 4);
  bf16_t* qrb  = (bf16_t*)alloc((size_t)L_SEQ * DMODEL * 2);
  bf16_t* krb  = (bf16_t*)alloc((size_t)L_SEQ * DMODEL * 2);
  bf16_t* qb   = (bf16_t*)alloc((size_t)L_SEQ * DMODEL * 2);
  bf16_t* kb   = (bf16_t*)alloc((size_t)L_SEQ * DMODEL * 2);
  bf16_t* attnb= (bf16_t*)alloc((size_t)NHEAD * L_SEQ * L_SEQ * 2);
  bf16_t* svT  = (bf16_t*)alloc((size_t)NHEAD * HDIM * L_SEQ * 2);
  float*  ASf  = (float*)alloc((size_t)L_SEQ * DMODEL * 4);
  float*  ARf  = (float*)alloc((size_t)L_SEQ * NHEAD * NRH * 4);
  bf16_t* Yb   = (bf16_t*)alloc((size_t)L_SEQ * DMODEL * 2);

  // 1. converts
  k_f32_to_bf16<<<2048, 256, 0, stream>>>(x, xb, L_SEQ * DMODEL);
  k_f32_to_bf16<<<2048, 256, 0, stream>>>(sym, symb, L_SEQ * DMODEL);
  k_transpose_to_bf16<<<4096, 256, 0, stream>>>(wq, wqT, DMODEL, DMODEL);
  k_transpose_to_bf16<<<1024, 256, 0, stream>>>(wk, wkT, DMODEL, 256);
  k_transpose_to_bf16<<<4096, 256, 0, stream>>>(wqr, wqrT, DMODEL, DMODEL);
  k_transpose_to_bf16<<<4096, 256, 0, stream>>>(wkr, wkrT, DMODEL, DMODEL);
  k_transpose_to_bf16<<<1024, 256, 0, stream>>>(wv, wvT, DMODEL, 256);
  k_transpose_to_bf16<<<4096, 256, 0, stream>>>(wo, woT, DMODEL, DMODEL);

  // 2. projections (WMMA bf16)
  k_gemm_bf16<<<dim3(32, 16, 1), 32, 0, stream>>>(xb, DMODEL, 0, wqT, DMODEL, 0,
      qf, nullptr, DMODEL, 1, 0, DMODEL, 1.0f);
  k_gemm_bf16<<<dim3(8, 16, 1), 32, 0, stream>>>(xb, DMODEL, 0, wkT, DMODEL, 0,
      kf, nullptr, 256, 1, 0, DMODEL, 1.0f);
  k_gemm_bf16<<<dim3(32, 16, 1), 32, 0, stream>>>(xb, DMODEL, 0, wqrT, DMODEL, 0,
      nullptr, qrb, DMODEL, 1, 0, DMODEL, 1.0f);
  k_gemm_bf16<<<dim3(32, 16, 1), 32, 0, stream>>>(xb, DMODEL, 0, wkrT, DMODEL, 0,
      nullptr, krb, DMODEL, 1, 0, DMODEL, 1.0f);
  k_gemm_bf16<<<dim3(8, 16, 1), 32, 0, stream>>>(symb, DMODEL, 0, wvT, DMODEL, 0,
      svf, nullptr, 256, 1, 0, DMODEL, 1.0f);

  // 3. RoPE + GQA replicate + V transpose
  k_rope_q<<<1024, 256, 0, stream>>>(qf, fc, fs, qb);
  k_rope_k_rep<<<256, 256, 0, stream>>>(kf, fc, fs, kb);
  k_svt_rep<<<2048, 256, 0, stream>>>(svf, svT);

  // 4. scores = scale * q k^T per head -> d_out attn region (pre-softmax)
  k_gemm_bf16<<<dim3(16, 16, NHEAD), 32, 0, stream>>>(qb, DMODEL, HDIM, kb, DMODEL, HDIM,
      attn, nullptr, L_SEQ, 1, (long)L_SEQ * L_SEQ, HDIM, ATTN_SCALE);

  // 5. causal softmax in place (+ bf16 copy)
  k_softmax<<<NHEAD * L_SEQ, 256, 0, stream>>>(attn, attnb);

  // 6. relations = scale * qr kr^T per relation head -> d_out [i][j][r]
  k_gemm_bf16<<<dim3(16, 16, NRH), 32, 0, stream>>>(qrb, DMODEL, HDIM, krb, DMODEL, HDIM,
      rel, nullptr, (long)L_SEQ * NRH, NRH, 1, HDIM, REL_SCALE);

  // 7. AR[i][h][r] = sum_j attn * relations
  k_ar<<<NHEAD * L_SEQ, 256, 0, stream>>>(attn, rel, ARf);

  // 8. attended_symbols = attn @ sv  -> AS [i][h*64+d]
  k_gemm_bf16<<<dim3(2, 16, NHEAD), 32, 0, stream>>>(attnb, L_SEQ, (long)L_SEQ * L_SEQ,
      svT, L_SEQ, (long)HDIM * L_SEQ,
      ASf, nullptr, DMODEL, 1, HDIM, L_SEQ, 1.0f);

  // 9. Y = AS + AR @ wr  (bf16)
  k_epilogue<<<2048, 256, 0, stream>>>(ASf, ARf, wr, Yb);

  // 10. out = Y @ wo
  k_gemm_bf16<<<dim3(32, 16, 1), 32, 0, stream>>>(Yb, DMODEL, 0, woT, DMODEL, 0,
      out, nullptr, DMODEL, 1, 0, DMODEL, 1.0f);
}
